// BahdanauAttention_38044820308305
// MI455X (gfx1250) — compile-verified, matrix operand order checked by reading, not run on hardware
//
#include <hip/hip_runtime.h>
#include <hip/hip_bf16.h>
#include <math.h>

// ---- problem constants -----------------------------------------------------
#define BB 64
#define TT 2048
#define KD 512
#define VD 512
#define AD 256

typedef __attribute__((ext_vector_type(16))) __bf16 v16bf;
typedef __attribute__((ext_vector_type(8)))  float  v8f;

union BfVec {
    unsigned short s[16];
    unsigned int   u[8];
    uint4          q[2];
    v16bf          v;
};

__device__ __forceinline__ unsigned int f32_to_bf16_rne(float f) {
    unsigned int x = __float_as_uint(f);
    x = x + 0x7FFFu + ((x >> 16) & 1u);   // round to nearest even
    return x >> 16;
}
__device__ __forceinline__ unsigned int pack2bf(float a, float b) {
    return f32_to_bf16_rne(a) | (f32_to_bf16_rne(b) << 16);
}

// ---- kernel 1: q_proj[b][a] = query[b]@Wq + bq[a] + bk[a] ------------------
__global__ void qproj_kernel(const float* __restrict__ query,
                             const float* __restrict__ Wq,
                             const float* __restrict__ bq,
                             const float* __restrict__ bk,
                             float* __restrict__ qp) {
    const int b = blockIdx.x;
    const int a = threadIdx.x;                 // 0..AD-1
    const float* q = query + (size_t)b * KD;
    float s = bq[a] + bk[a];
    #pragma unroll 8
    for (int k = 0; k < KD; ++k)
        s = fmaf(q[k], Wq[(size_t)k * AD + a], s);
    qp[(size_t)b * AD + a] = s;
}

// ---- kernel 2: fused scores = tanh(key@Wk_bf16 + qp)@Wo + bo ---------------
// 256 threads = 8 waves. 256KB LDS -> exactly 1 WG per WGP -> 2 waves/SIMD,
// so let the register allocator use up to half the file (no spills).
// Each wave owns 16-row tiles of the flattened [B*T, KD] key matrix; Wk is
// staged once per WG into LDS as bf16, pre-swizzled into B-fragment layout.
__global__ __launch_bounds__(256)
__attribute__((amdgpu_waves_per_eu(2, 2)))
void scores_kernel(const float* __restrict__ key,
                   const float* __restrict__ Wk,
                   const float* __restrict__ qp,
                   const float* __restrict__ Wo,
                   const float* __restrict__ bo,
                   float* __restrict__ scores) {
    extern __shared__ unsigned int ldsW[];     // 65536 uints = 256 KB
    const int tid  = threadIdx.x;
    const int lane = tid & 31;
    const int wave = tid >> 5;

    // ---- stage Wk (fp32 global -> bf16 LDS, fragment-swizzled) ----
    // fragment f = kcIdx*16 + nt  (kcIdx: K/32 chunk, nt: N/16 tile)
    // within fragment: lane-major, 8 uints (16 bf16) per lane
    for (int idx = tid; idx < (KD / 32) * (AD / 16) * 256; idx += blockDim.x) {
        const int frag  = idx >> 8;
        const int r     = idx & 255;
        const int l     = r >> 3;              // dest lane 0..31
        const int v     = r & 7;               // dest vgpr 0..7
        const int kcIdx = frag >> 4;
        const int nt    = frag & 15;
        const int n     = nt * 16 + (l & 15);
        int klo;
        if (l < 16) klo = (v < 4) ? (2 * v)     : (8 + 2 * v);   // K 0-7,16-23
        else        klo = (v < 4) ? (8 + 2 * v) : (16 + 2 * v);  // K 8-15,24-31
        const int k0 = kcIdx * 32 + klo;
        ldsW[idx] = pack2bf(Wk[(size_t)k0 * AD + n],
                            Wk[(size_t)(k0 + 1) * AD + n]);
    }
    __syncthreads();

    const int totalWaves = gridDim.x * (blockDim.x >> 5);
    const int gw         = blockIdx.x * (blockDim.x >> 5) + wave;
    const int nTiles     = (BB * TT) / 16;     // 8192
    const int kshift     = (lane >= 16) ? 8 : 0;
    const float bo0      = bo[0];

    for (int tile = gw; tile < nTiles; tile += totalWaves) {
        const int b     = tile >> 7;           // T/16 = 128 tiles per batch
        const int tbase = (tile & 127) << 4;
        const float* arow =
            key + ((size_t)b * TT + tbase + (lane & 15)) * KD + kshift;

        v8f acc[16] = {};                      // 16 N-tiles of 16x16 f32

        // software pipeline: preload first A chunk (this lane's 16 K-values)
        float4 f0 = *(const float4*)(arow + 0);
        float4 f1 = *(const float4*)(arow + 4);
        float4 f2 = *(const float4*)(arow + 16);
        float4 f3 = *(const float4*)(arow + 20);

        for (int kc = 0; kc < KD; kc += 32) {
            BfVec A;
            A.u[0] = pack2bf(f0.x, f0.y);  A.u[1] = pack2bf(f0.z, f0.w);
            A.u[2] = pack2bf(f1.x, f1.y);  A.u[3] = pack2bf(f1.z, f1.w);
            A.u[4] = pack2bf(f2.x, f2.y);  A.u[5] = pack2bf(f2.z, f2.w);
            A.u[6] = pack2bf(f3.x, f3.y);  A.u[7] = pack2bf(f3.z, f3.w);

            // issue next chunk's global loads before the WMMA chain
            if (kc + 32 < KD) {
                const float* nxt = arow + kc + 32;
                f0 = *(const float4*)(nxt + 0);
                f1 = *(const float4*)(nxt + 4);
                f2 = *(const float4*)(nxt + 16);
                f3 = *(const float4*)(nxt + 20);
            }

            const unsigned int* fb =
                &ldsW[(kc >> 5) * 16 * 256 + lane * 8];   // nt stride = 256

            // ping-pong B-fragment buffers: load nt+1 before consuming nt
            BfVec Bb[2];
            Bb[0].q[0] = ((const uint4*)fb)[0];
            Bb[0].q[1] = ((const uint4*)fb)[1];
            #pragma unroll
            for (int nt = 0; nt < 16; ++nt) {
                if (nt + 1 < 16) {
                    const uint4* pn = (const uint4*)(fb + (nt + 1) * 256);
                    Bb[(nt + 1) & 1].q[0] = pn[0];
                    Bb[(nt + 1) & 1].q[1] = pn[1];
                }
                acc[nt] = __builtin_amdgcn_wmma_f32_16x16x32_bf16(
                    false, A.v, false, Bb[nt & 1].v, (short)0, acc[nt],
                    false, false);
            }
        }

        // ---- epilogue: +qp, tanh, dot with Wo, reduce across 16 lanes ----
        const float* qpb = qp + (size_t)b * AD;
        float part[8];
        #pragma unroll
        for (int r = 0; r < 8; ++r) part[r] = 0.0f;

        #pragma unroll
        for (int nt = 0; nt < 16; ++nt) {
            const int n  = nt * 16 + (lane & 15);
            const float qv = qpb[n];
            const float wv = Wo[n];
            #pragma unroll
            for (int r = 0; r < 8; ++r)
                part[r] = fmaf(tanhf(acc[nt][r] + qv), wv, part[r]);
        }
        // lanes 0-15 hold rows 0..7 (vgpr r), lanes 16-31 rows 8..15
        #pragma unroll
        for (int r = 0; r < 8; ++r) {
            float p = part[r];
            #pragma unroll
            for (int m = 8; m >= 1; m >>= 1)
                p += __shfl_xor(p, m, 16);
            part[r] = p;
        }
        if ((lane & 15) == 0) {
            const int rowoff = (lane >> 4) * 8;
            #pragma unroll
            for (int r = 0; r < 8; ++r)
                scores[(size_t)b * TT + tbase + rowoff + r] = part[r] + bo0;
        }
    }
}

// ---- kernel 3: softmax over T per batch; writes attn into d_out ------------
__global__ void softmax_kernel(const float* __restrict__ scores,
                               float* __restrict__ attn) {
    __shared__ float red[256];
    const int b   = blockIdx.x;
    const int tid = threadIdx.x;
    const float* s = scores + (size_t)b * TT;

    float vals[TT / 256];
    float vmax = -INFINITY;
    #pragma unroll
    for (int i = 0; i < TT / 256; ++i) {
        vals[i] = s[tid + i * 256];
        vmax = fmaxf(vmax, vals[i]);
    }
    red[tid] = vmax;
    __syncthreads();
    for (int off = 128; off > 0; off >>= 1) {
        if (tid < off) red[tid] = fmaxf(red[tid], red[tid + off]);
        __syncthreads();
    }
    vmax = red[0];
    __syncthreads();

    float sum = 0.0f;
    #pragma unroll
    for (int i = 0; i < TT / 256; ++i) {
        vals[i] = __expf(vals[i] - vmax);
        sum += vals[i];
    }
    red[tid] = sum;
    __syncthreads();
    for (int off = 128; off > 0; off >>= 1) {
        if (tid < off) red[tid] += red[tid + off];
        __syncthreads();
    }
    const float inv = 1.0f / red[0];
    #pragma unroll
    for (int i = 0; i < TT / 256; ++i)
        attn[(size_t)b * TT + tid + i * 256] = vals[i] * inv;
}

// ---- kernel 4: context[b][v] = sum_t attn[b][t] * value[b][t][v] -----------
// grid = B * (VD/64); block 256 = 64 v-lanes x 4 t-groups
__global__ void context_kernel(const float* __restrict__ value,
                               const float* __restrict__ attn,
                               float* __restrict__ ctx) {
    __shared__ float red[256];
    const int b   = blockIdx.x >> 3;           // VD/64 = 8 chunks
    const int c   = blockIdx.x & 7;
    const int tid = threadIdx.x;
    const int v   = c * 64 + (tid & 63);
    const int tg  = tid >> 6;                  // 0..3

    const float* a  = attn + (size_t)b * TT;
    const float* vp = value + ((size_t)b * TT) * VD + v;

    float acc = 0.0f;
    for (int t = tg; t < TT; t += 4)
        acc = fmaf(a[t], vp[(size_t)t * VD], acc);

    red[tid] = acc;
    __syncthreads();
    if (tg == 0)
        ctx[(size_t)b * VD + v] =
            red[tid] + red[tid + 64] + red[tid + 128] + red[tid + 192];
}

// ---- host-side launcher ----------------------------------------------------
extern "C" void kernel_launch(void* const* d_in, const int* in_sizes, int n_in,
                              void* d_out, int out_size, void* d_ws, size_t ws_size,
                              hipStream_t stream) {
    const float* query = (const float*)d_in[0];   // [B,QD]
    const float* key   = (const float*)d_in[1];   // [B,T,KD]
    const float* value = (const float*)d_in[2];   // [B,T,VD]
    const float* Wq    = (const float*)d_in[3];   // [QD,AD]
    const float* bq    = (const float*)d_in[4];   // [AD]
    const float* Wk    = (const float*)d_in[5];   // [KD,AD]
    const float* bk    = (const float*)d_in[6];   // [AD]
    const float* Wo    = (const float*)d_in[7];   // [AD,1]
    const float* bo    = (const float*)d_in[8];   // [1]

    float* out_ctx  = (float*)d_out;                         // [B,VD]
    float* out_attn = (float*)d_out + (size_t)BB * VD;       // [B,T]

    float* ws_qp     = (float*)d_ws;                         // B*AD floats
    float* ws_scores = (float*)d_ws + (size_t)BB * AD;       // B*T floats

    // 1) q_proj (+bq +bk folded)
    qproj_kernel<<<BB, AD, 0, stream>>>(query, Wq, bq, bk, ws_qp);

    // 2) fused bf16-WMMA scores (Wk staged in 256 KB dynamic LDS)
    scores_kernel<<<512, 256, 256 * 1024, stream>>>(key, Wk, ws_qp, Wo, bo,
                                                    ws_scores);

    // 3) softmax over keys -> attn (second output)
    softmax_kernel<<<BB, 256, 0, stream>>>(ws_scores, out_attn);

    // 4) context = attn @ value (first output)
    context_kernel<<<BB * (VD / 64), 256, 0, stream>>>(value, out_attn, out_ctx);
}
